// RAILTDecoder_84997402788263
// MI455X (gfx1250) — compile-verified, matrix operand order checked by reading, not run on hardware
//
#include <hip/hip_runtime.h>

// ---------------------------------------------------------------------------
// Types for CDNA5 WMMA (wave32, 16x16x32 bf16 -> f32)
// ---------------------------------------------------------------------------
typedef __bf16 bf16_t;
typedef __attribute__((ext_vector_type(8)))  bf16_t v8bf;
typedef __attribute__((ext_vector_type(16))) bf16_t v16bf;
typedef __attribute__((ext_vector_type(8)))  float  v8f;
typedef __attribute__((ext_vector_type(4)))  unsigned int u32x4;
typedef __attribute__((ext_vector_type(8)))  int i32x8;
typedef __attribute__((ext_vector_type(4)))  int i32x4;

union FragU { v16bf v; v8bf h[2]; };
union Chunk8 { uint4 u; bf16_t h[8]; };

// Load a 16-lane-half A/B fragment: two contiguous 8-half (16B) LDS chunks,
// second chunk 16 halves after the first (CDNA5 16-bit A/B layout:
// lanes 0-15 K {0..7,16..23}, lanes 16-31 K {8..15,24..31}).
__device__ __forceinline__ v16bf ld_frag(const bf16_t* p) {
  FragU f;
  f.h[0] = *(const v8bf*)(p);
  f.h[1] = *(const v8bf*)(p + 16);
  return f.v;
}

#define NEG_INF (-__builtin_inff())

// ---------------------------------------------------------------------------
// TDM: one descriptor moves a 128-row x 32-half bf16 tile (row pitch K halves)
// from global memory into LDS, padding each 64B row with 16B so LDS rows land
// at a 40-half stride (bank-conflict-avoiding, matches ld_frag layout).
// D# packing per CDNA5 ISA 8.3/8.4: count=1, type=2, data_size=1 (2B),
// pad_enable, pad_interval code 3 (16 DWORDs), pad_amount code 3 (4 DWORDs),
// tensor_dim0=K, tensor_dim1=128, tile_dim0=32, tile_dim1=128, dim0_stride=K.
// workgroup_mask=0 (not in a cluster).
// This toolchain exposes the 6-arg builtin form:
//   (u32x4 g0, i32x8 g1, i32x4 g2, i32x4 g3, i32x8, i32 cpol)
// ---------------------------------------------------------------------------
__device__ __forceinline__ void tdm_load_tile(unsigned lds_addr,
                                              const bf16_t* gsrc, int K)
{
  unsigned long long ga = (unsigned long long)(uintptr_t)gsrc;
  u32x4 g0;
  g0[0] = 1u;                                          // count=1, user mode
  g0[1] = lds_addr;                                    // lds_addr
  g0[2] = (unsigned)(ga & 0xffffffffu);                // global_addr[31:0]
  g0[3] = (unsigned)((ga >> 32) & 0x01ffffffu)         // global_addr[56:32]
        | 0x80000000u;                                 // type=2
  i32x8 g1;
  g1[0] = (1 << 16)                                    // data_size = 2 bytes
        | (1 << 20)                                    // pad_enable
        | (3 << 22)                                    // pad_interval: 16 DWORDs
        | (3 << 25);                                   // pad_amount: 4 DWORDs
  g1[1] = (int)(((unsigned)K & 0xffffu) << 16);        // tensor_dim0 lo16
  g1[2] = (int)((((unsigned)K >> 16) & 0xffffu)        // tensor_dim0 hi16
        | (128u << 16));                               // tensor_dim1 lo16
  g1[3] = (int)(32u << 16);                            // dim1 hi=0 | tile_dim0=32
  g1[4] = 128;                                         // tile_dim1=128, tile_dim2=0
  g1[5] = K;                                           // tensor_dim0_stride lo32
  g1[6] = 0;                                           // stride hi | dim1_stride lo
  g1[7] = 0;
  i32x4 gz4;
  gz4[0] = 0; gz4[1] = 0; gz4[2] = 0; gz4[3] = 0;      // 2D: groups 2/3 unused
  i32x8 gz8;
  gz8[0] = 0; gz8[1] = 0; gz8[2] = 0; gz8[3] = 0;
  gz8[4] = 0; gz8[5] = 0; gz8[6] = 0; gz8[7] = 0;
  __builtin_amdgcn_tensor_load_to_lds(g0, g1, gz4, gz4, gz8, 0);
}

// ---------------------------------------------------------------------------
// Tiled WMMA GEMM:  out = act(A @ W^T + bias) [+ res]
//   A: (Mrows, K) bf16 contiguous; W: (N, K) bf16 row-major
//   out/res: row pitch N, rows mapped via (rpbO,bsO)/(rpbR,bsR)
// Block: 256 threads = 8 waves (4x2), block tile 128x128, wave tile 32x64,
// K-step 32, double-buffered LDS filled by the Tensor Data Mover
// (TENSORcnt-tracked DMA issued by wave 0; two descriptors per K-tile).
// ---------------------------------------------------------------------------
template<int ACT, bool WF, bool WB, bool RES>
__global__ __launch_bounds__(256)
void wmma_gemm(const bf16_t* __restrict__ Ag,
               const bf16_t* __restrict__ Wg, const float* __restrict__ bias,
               const float* __restrict__ resp, int rpbR, long long bsR,
               float* __restrict__ outF, bf16_t* __restrict__ outB,
               int rpbO, long long bsO, int N, int K)
{
  __shared__ bf16_t As[2][128 * 40];   // 128 x 32, padded stride 40 halves
  __shared__ bf16_t Bs[2][128 * 40];

  const int tid  = threadIdx.x;
  const int lane = tid & 31, wave = tid >> 5;
  const int lr = lane & 15, hi = lane >> 4;
  const int wm = wave >> 1, wn = wave & 1;        // 4x2 wave grid
  const int bm0 = blockIdx.y * 128, bn0 = blockIdx.x * 128;

  const unsigned asA[2] = {(unsigned)(uintptr_t)&As[0][0],
                           (unsigned)(uintptr_t)&As[1][0]};
  const unsigned asB[2] = {(unsigned)(uintptr_t)&Bs[0][0],
                           (unsigned)(uintptr_t)&Bs[1][0]};
  const bf16_t* aBase = Ag + (long long)bm0 * K;
  const bf16_t* wBase = Wg + (long long)bn0 * K;

  const v8f vz = {0.f,0.f,0.f,0.f,0.f,0.f,0.f,0.f};
  v8f acc[2][4];
#pragma unroll
  for (int mi = 0; mi < 2; ++mi)
#pragma unroll
    for (int ni = 0; ni < 4; ++ni) acc[mi][ni] = vz;

  if (wave == 0) {
    tdm_load_tile(asA[0], aBase, K);
    tdm_load_tile(asB[0], wBase, K);
  }
  int bsel = 0;
  for (int k0 = 0; k0 < K; k0 += 32) {
    if (wave == 0) {
      if (k0 + 32 < K) {
        tdm_load_tile(asA[bsel ^ 1], aBase + k0 + 32, K);   // prefetch next tile
        tdm_load_tile(asB[bsel ^ 1], wBase + k0 + 32, K);
        __builtin_amdgcn_s_wait_tensorcnt(2);               // current tile landed
      } else {
        __builtin_amdgcn_s_wait_tensorcnt(0);
      }
    }
    __syncthreads();

    v16bf af[2], bfv[4];
#pragma unroll
    for (int mi = 0; mi < 2; ++mi)
      af[mi] = ld_frag(&As[bsel][(wm * 32 + mi * 16 + lr) * 40 + hi * 8]);
#pragma unroll
    for (int ni = 0; ni < 4; ++ni)
      bfv[ni] = ld_frag(&Bs[bsel][(wn * 64 + ni * 16 + lr) * 40 + hi * 8]);
#pragma unroll
    for (int mi = 0; mi < 2; ++mi)
#pragma unroll
      for (int ni = 0; ni < 4; ++ni)
        acc[mi][ni] = __builtin_amdgcn_wmma_f32_16x16x32_bf16(
            false, af[mi], false, bfv[ni], (short)0, acc[mi][ni], false, false);

    __syncthreads();                                // buffer reuse fence
    bsel ^= 1;
  }

  // epilogue: C layout -> lane holds col = lr, rows r (+8 for hi lanes)
#pragma unroll
  for (int mi = 0; mi < 2; ++mi)
#pragma unroll
    for (int ni = 0; ni < 4; ++ni)
#pragma unroll
      for (int r = 0; r < 8; ++r) {
        int grow = bm0 + wm * 32 + mi * 16 + r + hi * 8;
        int gcol = bn0 + wn * 64 + ni * 16 + lr;
        float v = acc[mi][ni][r] + bias[gcol];
        if (ACT == 1) v = v > 0.f ? v : 0.f;
        if (ACT == 2) v = 1.f / (1.f + __expf(-v));
        if (RES) {
          int bb = grow / rpbR, tt = grow - bb * rpbR;
          v += resp[(long long)bb * bsR + (long long)tt * N + gcol];
        }
        int bo = grow / rpbO, to = grow - bo * rpbO;
        long long oidx = (long long)bo * bsO + (long long)to * N + gcol;
        if (WF) outF[oidx] = v;
        if (WB) outB[oidx] = (bf16_t)v;
      }
}

// ---------------------------------------------------------------------------
// Flash attention per (b,h). Lq<=144, Lk<=144, DH=64, scale 1/8.
// maskMode 0 (self): mask j>=16 && q>=16 && (j-16)>(q-16)
// maskMode 1 (cross): mask j>=16 && (j-16)>q
// Block = 256 threads (8 waves); each wave owns 16 query rows (grid-stride).
// Q/K staged with GLOBAL_LOAD_ASYNC_TO_LDS_B128 (ASYNCcnt), V transposed
// manually (needs per-element scatter).
// ---------------------------------------------------------------------------
__global__ __launch_bounds__(256)
void attn_kernel(const bf16_t* __restrict__ Qb, long long qBS, int qRS,
                 const bf16_t* __restrict__ Kb, const bf16_t* __restrict__ Vb,
                 long long kvBS, int kvRS,
                 bf16_t* __restrict__ Out, long long oBS,
                 int Lq, int Lk, int maskMode)
{
  __shared__ bf16_t Qs[144 * 64];
  __shared__ bf16_t Ks[144 * 64];
  __shared__ bf16_t Vt[64 * 160];     // V transposed, keys padded to 160
  __shared__ bf16_t Pw[8 * 512];      // per-wave 16x32 P staging

  const int tid  = threadIdx.x;
  const int b = blockIdx.x >> 4, h = blockIdx.x & 15;
  const int lane = tid & 31, wave = tid >> 5;
  const int lr = lane & 15, hi = lane >> 4;

  for (int i = tid; i < Lq * 8; i += 256) {
    int q = i >> 3, c = (i & 7) * 8;
    unsigned ld = (unsigned)(uintptr_t)&Qs[q * 64 + c];
    unsigned long long ga = (unsigned long long)(uintptr_t)
        (Qb + (long long)b * qBS + (long long)q * qRS + h * 64 + c);
    asm volatile("global_load_async_to_lds_b128 %0, %1, off"
                 :: "v"(ld), "v"(ga) : "memory");
  }
  for (int i = tid; i < Lk * 8; i += 256) {
    int kk = i >> 3, c = (i & 7) * 8;
    unsigned ld = (unsigned)(uintptr_t)&Ks[kk * 64 + c];
    unsigned long long ga = (unsigned long long)(uintptr_t)
        (Kb + (long long)b * kvBS + (long long)kk * kvRS + h * 64 + c);
    asm volatile("global_load_async_to_lds_b128 %0, %1, off"
                 :: "v"(ld), "v"(ga) : "memory");
  }
  for (int i = tid; i < (64 * 160) / 8; i += 256) {
    uint4 zz; zz.x = 0u; zz.y = 0u; zz.z = 0u; zz.w = 0u;
    *(uint4*)&Vt[i * 8] = zz;
  }
  __syncthreads();
  for (int i = tid; i < Lk * 8; i += 256) {
    int kk = i >> 3, c = (i & 7) * 8;
    Chunk8 ch;
    ch.u = *(const uint4*)(Vb + (long long)b * kvBS + (long long)kk * kvRS + h * 64 + c);
#pragma unroll
    for (int j = 0; j < 8; ++j) Vt[(c + j) * 160 + kk] = ch.h[j];
  }
  asm volatile("s_wait_asynccnt 0" ::: "memory");   // Q/K tiles landed
  __syncthreads();

  const v8f vz = {0.f,0.f,0.f,0.f,0.f,0.f,0.f,0.f};
  const float scale = 0.125f;
  bf16_t* pw = &Pw[wave * 512];

  for (int qt = wave; qt * 16 < Lq; qt += 8) {
    v16bf qf[2];
#pragma unroll
    for (int s = 0; s < 2; ++s)
      qf[s] = ld_frag(&Qs[(qt * 16 + lr) * 64 + s * 32 + hi * 8]);

    float m[8], l[8];
    v8f o[4];
#pragma unroll
    for (int r = 0; r < 8; ++r) { m[r] = NEG_INF; l[r] = 0.f; }
    o[0] = vz; o[1] = vz; o[2] = vz; o[3] = vz;

    const int nPairs = (Lk + 31) >> 5;
    for (int kp = 0; kp < nPairs; ++kp) {
      float sv[2][8];
#pragma unroll
      for (int t = 0; t < 2; ++t) {
        int kc0 = (kp * 2 + t) * 16;
        if (kc0 < Lk) {
          v8f sa = vz;
#pragma unroll
          for (int s = 0; s < 2; ++s) {
            v16bf kf = ld_frag(&Ks[(kc0 + lr) * 64 + s * 32 + hi * 8]);
            sa = __builtin_amdgcn_wmma_f32_16x16x32_bf16(
                false, qf[s], false, kf, (short)0, sa, false, false);
          }
          int j = kc0 + lr;
#pragma unroll
          for (int r = 0; r < 8; ++r) {
            int q = qt * 16 + r + hi * 8;
            bool msk = (maskMode == 0)
                         ? (j >= 16 && q >= 16 && (j - 16) > (q - 16))
                         : (j >= 16 && (j - 16) > q);
            sv[t][r] = msk ? NEG_INF : sa[r] * scale;
          }
        } else {
#pragma unroll
          for (int r = 0; r < 8; ++r) sv[t][r] = NEG_INF;
        }
      }
      // online softmax per row (row = 16-lane half of the wave)
#pragma unroll
      for (int r = 0; r < 8; ++r) {
        float mx = fmaxf(sv[0][r], sv[1][r]);
        mx = fmaxf(mx, __shfl_xor(mx, 1));
        mx = fmaxf(mx, __shfl_xor(mx, 2));
        mx = fmaxf(mx, __shfl_xor(mx, 4));
        mx = fmaxf(mx, __shfl_xor(mx, 8));
        float mn = fmaxf(m[r], mx);
        float p0 = __expf(sv[0][r] - mn);
        float p1 = __expf(sv[1][r] - mn);
        float rs = p0 + p1;
        rs += __shfl_xor(rs, 1);
        rs += __shfl_xor(rs, 2);
        rs += __shfl_xor(rs, 4);
        rs += __shfl_xor(rs, 8);
        float al = __expf(m[r] - mn);
        l[r] = l[r] * al + rs;
        m[r] = mn;
        o[0][r] = o[0][r] * al; o[1][r] = o[1][r] * al;
        o[2][r] = o[2][r] * al; o[3][r] = o[3][r] * al;
        pw[(r + hi * 8) * 32 + lr]      = (bf16_t)p0;
        pw[(r + hi * 8) * 32 + 16 + lr] = (bf16_t)p1;
      }
      // P (C-layout -> A-layout via wave-private LDS), then P @ V
      v16bf pa = ld_frag(&pw[lr * 32 + hi * 8]);
#pragma unroll
      for (int nt = 0; nt < 4; ++nt) {
        v16bf vf = ld_frag(&Vt[(nt * 16 + lr) * 160 + kp * 32 + hi * 8]);
        o[nt] = __builtin_amdgcn_wmma_f32_16x16x32_bf16(
            false, pa, false, vf, (short)0, o[nt], false, false);
      }
    }
#pragma unroll
    for (int r = 0; r < 8; ++r) {
      float inv = 1.f / l[r];
      int q = qt * 16 + r + hi * 8;
      bf16_t* op = Out + (long long)b * oBS + (long long)q * 1024 + h * 64;
#pragma unroll
      for (int nt = 0; nt < 4; ++nt)
        op[nt * 16 + lr] = (bf16_t)(o[nt][r] * inv);
    }
  }
}

// ---------------------------------------------------------------------------
// LayerNorm over D=1024, one block per row, bf16 output.
// ---------------------------------------------------------------------------
__global__ __launch_bounds__(256)
void ln_kernel(const float* __restrict__ X, int rpb, long long bs,
               const float* __restrict__ g, const float* __restrict__ be,
               bf16_t* __restrict__ Y)
{
  int row = blockIdx.x;
  int b = row / rpb, t = row - b * rpb;
  const float* x = X + (long long)b * bs + (long long)t * 1024;
  int tid = threadIdx.x;
  float4 v = *(const float4*)(x + tid * 4);
  float s  = v.x + v.y + v.z + v.w;
  float s2 = v.x * v.x + v.y * v.y + v.z * v.z + v.w * v.w;
#pragma unroll
  for (int d = 1; d < 32; d <<= 1) { s += __shfl_xor(s, d); s2 += __shfl_xor(s2, d); }
  __shared__ float ss[8], ss2[8];
  if ((tid & 31) == 0) { ss[tid >> 5] = s; ss2[tid >> 5] = s2; }
  __syncthreads();
  float ts = 0.f, ts2 = 0.f;
#pragma unroll
  for (int w = 0; w < 8; ++w) { ts += ss[w]; ts2 += ss2[w]; }
  float mu  = ts * (1.f / 1024.f);
  float var = ts2 * (1.f / 1024.f) - mu * mu;
  float rs  = rsqrtf(var + 1e-5f);
  float4 gv = *(const float4*)(g + tid * 4);
  float4 bv = *(const float4*)(be + tid * 4);
  bf16_t* y = Y + (long long)row * 1024 + tid * 4;
  y[0] = (bf16_t)((v.x - mu) * rs * gv.x + bv.x);
  y[1] = (bf16_t)((v.y - mu) * rs * gv.y + bv.y);
  y[2] = (bf16_t)((v.z - mu) * rs * gv.z + bv.z);
  y[3] = (bf16_t)((v.w - mu) * rs * gv.w + bv.w);
}

// z[b,i,:] = (i<16 ? mem[b,i,:] : emb[in_res[b, kidx*128 + i-16]]) + pos[i,:]
__global__ void build_z_kernel(const float* __restrict__ mem, const int* __restrict__ in_res,
                               const float* __restrict__ emb, const float* __restrict__ pos,
                               float* __restrict__ z, int kidx)
{
  int row = blockIdx.x;               // B*144
  int b = row / 144, i = row - b * 144;
  int c = threadIdx.x * 4;
  float4 pv = *(const float4*)(pos + (long long)i * 1024 + c);
  float4 xv;
  if (i < 16) {
    xv = *(const float4*)(mem + (long long)(b * 16 + i) * 1024 + c);
  } else {
    int tok = in_res[b * 2048 + kidx * 128 + (i - 16)];
    xv = *(const float4*)(emb + (long long)tok * 1024 + c);
  }
  float4 ov; ov.x = xv.x + pv.x; ov.y = xv.y + pv.y; ov.z = xv.z + pv.z; ov.w = xv.w + pv.w;
  *(float4*)(z + (long long)row * 1024 + c) = ov;
}

// kvs[b,i,:] = i<16 ? proj_mem[b,i,:] : enc_out[b, kidx*128 + i-16, :]
__global__ void build_kv_kernel(const float* __restrict__ proj, const float* __restrict__ enc_out,
                                float* __restrict__ kvs, int kidx)
{
  int row = blockIdx.x;               // B*144
  int b = row / 144, i = row - b * 144;
  int c = threadIdx.x * 4;
  float4 v;
  if (i < 16)
    v = *(const float4*)(proj + (long long)(b * 16 + i) * 1024 + c);
  else
    v = *(const float4*)(enc_out + ((long long)b * 2048 + kidx * 128 + (i - 16)) * 1024 + c);
  *(float4*)(kvs + (long long)row * 1024 + c) = v;
}

// mem = kidx>0 ? gate*mem_cand + (1-gate)*mem : mem_cand    (mem_cand = z[:, :16])
__global__ void mem_update_kernel(const float* __restrict__ gate, const float* __restrict__ z,
                                  float* __restrict__ mem, int kidx)
{
  int row = blockIdx.x;               // B*16
  int b = row >> 4, i = row & 15;
  int c = threadIdx.x * 4;
  float4 mc = *(const float4*)(z + ((long long)b * 144 + i) * 1024 + c);
  float4 out;
  if (kidx > 0) {
    float4 gv = *(const float4*)(gate + (long long)row * 1024 + c);
    float4 mv = *(const float4*)(mem + (long long)row * 1024 + c);
    out.x = gv.x * mc.x + (1.f - gv.x) * mv.x;
    out.y = gv.y * mc.y + (1.f - gv.y) * mv.y;
    out.z = gv.z * mc.z + (1.f - gv.z) * mv.z;
    out.w = gv.w * mc.w + (1.f - gv.w) * mv.w;
  } else out = mc;
  *(float4*)(mem + (long long)row * 1024 + c) = out;
}

// f32 -> bf16 (strided rows), output contiguous
__global__ void cvt_kernel(const float* __restrict__ X, int rpb, long long bs, int rowlen,
                           bf16_t* __restrict__ Y)
{
  int row = blockIdx.x;
  int b = row / rpb, t = row - b * rpb;
  const float* x = X + (long long)b * bs + (long long)t * rowlen;
  bf16_t* y = Y + (long long)row * rowlen;
  for (int c = threadIdx.x * 4; c < rowlen; c += 1024) {
    float4 v = *(const float4*)(x + c);
    y[c]     = (bf16_t)v.x;
    y[c + 1] = (bf16_t)v.y;
    y[c + 2] = (bf16_t)v.z;
    y[c + 3] = (bf16_t)v.w;
  }
}

// ---------------------------------------------------------------------------
// Host: wire the whole decoder (16 sequential chunk steps).
// ---------------------------------------------------------------------------
template<int ACT, bool WF, bool WB, bool RES>
static inline void gemm(hipStream_t st, const bf16_t* A,
                        const bf16_t* W, const float* bias,
                        const float* res, int rpbR, long long bsR,
                        float* oF, bf16_t* oB, int rpbO, long long bsO,
                        int Mr, int N, int K)
{
  dim3 g(N / 128, Mr / 128);
  wmma_gemm<ACT, WF, WB, RES><<<g, dim3(256), 0, st>>>(
      A, W, bias, res, rpbR, bsR, oF, oB, rpbO, bsO, N, K);
}

extern "C" void kernel_launch(void* const* d_in, const int* in_sizes, int n_in,
                              void* d_out, int out_size, void* d_ws, size_t ws_size,
                              hipStream_t stream)
{
  (void)in_sizes; (void)n_in; (void)out_size; (void)ws_size;
  const int*   in_res  = (const int*)  d_in[0];
  const float* enc_out = (const float*)d_in[1];
  const float* enc_mem = (const float*)d_in[2];
  const float* emb     = (const float*)d_in[3];
  const float* Wmem    = (const float*)d_in[4];
  const float* bmem    = (const float*)d_in[5];
  const float* pos     = (const float*)d_in[6];
  const float* Wqkv_s  = (const float*)d_in[7];
  const float* bqkv_s  = (const float*)d_in[8];
  const float* Wo_s    = (const float*)d_in[9];
  const float* bo_s    = (const float*)d_in[10];
  const float* g_la    = (const float*)d_in[11];
  const float* b_la    = (const float*)d_in[12];
  const float* W1      = (const float*)d_in[13];
  const float* b1      = (const float*)d_in[14];
  const float* W2      = (const float*)d_in[15];
  const float* b2      = (const float*)d_in[16];
  const float* g_lf    = (const float*)d_in[17];
  const float* b_lf    = (const float*)d_in[18];
  const float* Wencm   = (const float*)d_in[19];
  const float* bencm   = (const float*)d_in[20];
  const float* Wqkv_c  = (const float*)d_in[21];
  const float* bqkv_c  = (const float*)d_in[22];
  const float* Wo_c    = (const float*)d_in[23];
  const float* bo_c    = (const float*)d_in[24];
  const float* g_lq    = (const float*)d_in[25];
  const float* b_lq    = (const float*)d_in[26];
  const float* g_lkv   = (const float*)d_in[27];
  const float* b_lkv   = (const float*)d_in[28];
  const float* W1c     = (const float*)d_in[29];
  const float* b1c     = (const float*)d_in[30];
  const float* W2c     = (const float*)d_in[31];
  const float* b2c     = (const float*)d_in[32];
  const float* g_lfc   = (const float*)d_in[33];
  const float* b_lfc   = (const float*)d_in[34];
  const float* Wg      = (const float*)d_in[35];
  const float* bg      = (const float*)d_in[36];
  float* out = (float*)d_out;

  char* ws = (char*)d_ws;
  size_t off = 0;
  auto alloc = [&](size_t bytes) -> char* {
    char* p = ws + off;
    off = (off + bytes + 255) & ~(size_t)255;
    return p;
  };
  auto abf  = [&](size_t n) -> bf16_t* { return (bf16_t*)alloc(n * sizeof(bf16_t)); };
  auto af32 = [&](size_t n) -> float*  { return (float*)alloc(n * sizeof(float)); };

  const size_t DD = 1024ull * 1024, DDF = 1024ull * 4096;
  const int Brows = 16 * 144;   // 2304
  const int Hrows = 16 * 128;   // 2048

  bf16_t* wb_mem  = abf(DD);
  bf16_t* wb_encm = abf(DD);
  bf16_t* wb_qkvs = abf(3 * DD);
  bf16_t* wb_wos  = abf(DD);
  bf16_t* wb_w1   = abf(DDF);
  bf16_t* wb_w2   = abf(DDF);
  bf16_t* wb_qkvc = abf(3 * DD);
  bf16_t* wb_woc  = abf(DD);
  bf16_t* wb_w1c  = abf(DDF);
  bf16_t* wb_w2c  = abf(DDF);
  bf16_t* wb_wg   = abf(DD);
  bf16_t* encm_b  = abf(256ull * 1024);

  float*  z    = af32((size_t)Brows * 1024);
  bf16_t* Xb   = abf((size_t)Brows * 1024);
  bf16_t* Yb   = abf((size_t)Brows * 1024);
  bf16_t* QKV  = abf((size_t)Brows * 3072);
  bf16_t* H1   = abf((size_t)Brows * 4096);
  float*  KVS  = af32((size_t)Brows * 1024);
  bf16_t* QC   = abf((size_t)Hrows * 1024);
  bf16_t* KVC  = abf((size_t)Brows * 2048);
  float*  memb = af32(256ull * 1024);
  float*  proj = af32(256ull * 1024);
  float*  gate = af32(256ull * 1024);
  bf16_t* mcb  = abf(256ull * 1024);

  // ---- weight / input conversions to bf16 (every call; deterministic) ----
  cvt_kernel<<<1024, 256, 0, stream>>>(Wmem,   1024, (long long)DD,    1024, wb_mem);
  cvt_kernel<<<1024, 256, 0, stream>>>(Wencm,  1024, (long long)DD,    1024, wb_encm);
  cvt_kernel<<<3072, 256, 0, stream>>>(Wqkv_s, 3072, (long long)(3*DD),1024, wb_qkvs);
  cvt_kernel<<<1024, 256, 0, stream>>>(Wo_s,   1024, (long long)DD,    1024, wb_wos);
  cvt_kernel<<<4096, 256, 0, stream>>>(W1,     4096, (long long)DDF,   1024, wb_w1);
  cvt_kernel<<<1024, 256, 0, stream>>>(W2,     1024, (long long)DDF,   4096, wb_w2);
  cvt_kernel<<<3072, 256, 0, stream>>>(Wqkv_c, 3072, (long long)(3*DD),1024, wb_qkvc);
  cvt_kernel<<<1024, 256, 0, stream>>>(Wo_c,   1024, (long long)DD,    1024, wb_woc);
  cvt_kernel<<<4096, 256, 0, stream>>>(W1c,    4096, (long long)DDF,   1024, wb_w1c);
  cvt_kernel<<<1024, 256, 0, stream>>>(W2c,    1024, (long long)DDF,   4096, wb_w2c);
  cvt_kernel<<<1024, 256, 0, stream>>>(Wg,     1024, (long long)DD,    1024, wb_wg);
  cvt_kernel<<<256,  256, 0, stream>>>(enc_mem, 256, 256ll * 1024,     1024, encm_b);

  // ---- mem0 / proj_mem ----
  gemm<0,true,false,false>(stream, encm_b, wb_mem,  bmem,  nullptr, 1, 0,
                           memb, nullptr, 256, 0, 256, 1024, 1024);
  gemm<0,true,false,false>(stream, encm_b, wb_encm, bencm, nullptr, 1, 0,
                           proj, nullptr, 256, 0, 256, 1024, 1024);

  const long long zh_bs = 144ll * 1024;   // batch stride of z rows
  float* zh = z + 16 * 1024;              // h = z[:, 16:, :]

  for (int kidx = 0; kidx < 16; ++kidx) {
    // z = concat(mem, emb[chunk]) + pos
    build_z_kernel<<<Brows, 256, 0, stream>>>(memb, in_res, emb, pos, z, kidx);
    // ---- self block ----
    ln_kernel<<<Brows, 256, 0, stream>>>(z, Brows, 0, g_la, b_la, Xb);
    gemm<0,false,true,false>(stream, Xb, wb_qkvs, bqkv_s, nullptr, 1, 0,
                             nullptr, QKV, Brows, 0, Brows, 3072, 1024);
    attn_kernel<<<256, 256, 0, stream>>>(QKV, 144ll * 3072, 3072,
                                         QKV + 1024, QKV + 2048, 144ll * 3072, 3072,
                                         Yb, 144ll * 1024, 144, 144, 0);
    gemm<0,true,false,true>(stream, Yb, wb_wos, bo_s, z, Brows, 0,
                            z, nullptr, Brows, 0, Brows, 1024, 1024);
    ln_kernel<<<Brows, 256, 0, stream>>>(z, Brows, 0, g_lf, b_lf, Xb);
    gemm<1,false,true,false>(stream, Xb, wb_w1, b1, nullptr, 1, 0,
                             nullptr, H1, Brows, 0, Brows, 4096, 1024);
    gemm<0,true,false,true>(stream, H1, wb_w2, b2, z, Brows, 0,
                            z, nullptr, Brows, 0, Brows, 1024, 4096);
    // ---- cross block (h = z[:,16:], mem_cand = z[:,:16] stays untouched) ----
    ln_kernel<<<Hrows, 256, 0, stream>>>(zh, 128, zh_bs, g_lq, b_lq, Xb);
    build_kv_kernel<<<Brows, 256, 0, stream>>>(proj, enc_out, KVS, kidx);
    ln_kernel<<<Brows, 256, 0, stream>>>(KVS, Brows, 0, g_lkv, b_lkv, Yb);
    gemm<0,false,true,false>(stream, Xb, wb_qkvc, bqkv_c, nullptr, 1, 0,
                             nullptr, QC, Hrows, 0, Hrows, 1024, 1024);
    gemm<0,false,true,false>(stream, Yb, wb_qkvc + DD, bqkv_c + 1024,
                             nullptr, 1, 0, nullptr, KVC, Brows, 0, Brows, 2048, 1024);
    attn_kernel<<<256, 256, 0, stream>>>(QC, 128ll * 1024, 1024,
                                         KVC, KVC + 1024, 144ll * 2048, 2048,
                                         Yb, 128ll * 1024, 128, 144, 1);
    gemm<0,true,false,true>(stream, Yb, wb_woc, bo_c, zh, 128, zh_bs,
                            zh, nullptr, 128, zh_bs, Hrows, 1024, 1024);
    ln_kernel<<<Hrows, 256, 0, stream>>>(zh, 128, zh_bs, g_lfc, b_lfc, Xb);
    gemm<1,false,true,false>(stream, Xb, wb_w1c, b1c, nullptr, 1, 0,
                             nullptr, H1, Hrows, 0, Hrows, 4096, 1024);
    // final residual FFN writes directly into d_out[b, kidx*128 + t, :]
    gemm<0,true,false,true>(stream, H1, wb_w2c, b2c, zh, 128, zh_bs,
                            out + (long long)kidx * 128 * 1024, nullptr,
                            128, 2048ll * 1024, Hrows, 1024, 4096);
    // ---- gated memory update ----
    cvt_kernel<<<256, 256, 0, stream>>>(z, 16, zh_bs, 1024, mcb);
    gemm<2,true,false,false>(stream, mcb, wb_wg, bg, nullptr, 1, 0,
                             gate, nullptr, 256, 0, 256, 1024, 1024);
    mem_update_kernel<<<256, 256, 0, stream>>>(gate, z, memb, kidx);
  }
}
